// BiLSTM_CRF_60722247631040
// MI455X (gfx1250) — matrix operand order repaired and checked by reading, compiled, not run on hardware
//
#include <hip/hip_runtime.h>
#include <hip/hip_bf16.h>

// ---------------- problem constants (from reference) ----------------
constexpr int TT   = 2048;   // sentence length
constexpr int DD   = 300;    // embedding dim
constexpr int DP   = 320;    // embedding dim padded to multiple of 32 (WMMA K)
constexpr int HH   = 256;    // per-direction hidden
constexpr int GG   = 1024;   // 4*H gate width
constexpr int HIDN = 512;    // 2*H
constexpr int KK   = 48;     // tagset
constexpr int KSTART = 46;
constexpr int KSTOP  = 47;

typedef __attribute__((ext_vector_type(16))) __bf16 v16bf;
typedef __attribute__((ext_vector_type(8)))  float  v8f;

__device__ __forceinline__ float sigmoidf_(float x) {
    return 1.0f / (1.0f + __expf(-x));
}
__device__ __forceinline__ unsigned short f32_to_bf16_bits(float f) {
    return (unsigned short)(__float_as_uint(f) >> 16);
}

// ---------------- prep: gather embeddings -> bf16, pad D to 320 ----------------
__global__ void prep_x_kernel(const int* __restrict__ sent,
                              const float* __restrict__ E,
                              unsigned short* __restrict__ xbf) {
    int i = blockIdx.x * blockDim.x + threadIdx.x;
    if (i >= TT * DP) return;
    int t = i / DP, k = i - t * DP;
    float v = (k < DD) ? E[(size_t)sent[t] * DD + k] : 0.0f;
    xbf[i] = f32_to_bf16_bits(v);
}

// ---------------- prep: W_ih (both dirs) -> bf16, pad D ----------------
__global__ void prep_wih_kernel(const float* __restrict__ wf,
                                const float* __restrict__ wb,
                                unsigned short* __restrict__ wbf) {
    int i = blockIdx.x * blockDim.x + threadIdx.x;
    if (i >= 2 * GG * DP) return;
    int dir = i / (GG * DP);
    int r   = (i / DP) % GG;
    int k   = i % DP;
    const float* src = dir ? wb : wf;
    float v = (k < DD) ? src[(size_t)r * DD + k] : 0.0f;
    wbf[i] = f32_to_bf16_bits(v);
}

// ---------------- prep: W_hh (both dirs) -> packed bf16x2 ----------------
__global__ void prep_whh_kernel(const float* __restrict__ wf,
                                const float* __restrict__ wb,
                                unsigned int* __restrict__ wpk) {
    int i = blockIdx.x * blockDim.x + threadIdx.x;
    if (i >= 2 * GG * (HH / 2)) return;
    int dir = i / (GG * (HH / 2));
    int r   = (i / (HH / 2)) % GG;
    int j   = i % (HH / 2);
    const float* src = dir ? wb : wf;
    unsigned int lo = f32_to_bf16_bits(src[(size_t)r * HH + 2 * j]);
    unsigned int hi = f32_to_bf16_bits(src[(size_t)r * HH + 2 * j + 1]);
    wpk[i] = lo | (hi << 16);
}

// ---------------- input projection: xg = x @ W_ih^T + b_ih via WMMA ----------------
// Each wave owns one 16x16 output tile; k swept in 32-wide bf16 WMMA steps.
__global__ void __launch_bounds__(128)
xg_gemm_kernel(const __bf16* __restrict__ xbf,        // [TT][DP]
               const __bf16* __restrict__ wf,         // [GG][DP]
               const __bf16* __restrict__ wb,         // [GG][DP]
               const float* __restrict__ bih_f,
               const float* __restrict__ bih_b,
               float* __restrict__ xg_f,              // [TT][GG]
               float* __restrict__ xg_b) {
    const int lane = threadIdx.x & 31;
    const int wave = threadIdx.x >> 5;
    const int m0   = blockIdx.x * 16;
    const int n0   = (blockIdx.y * 4 + wave) * 16;
    const int dir  = blockIdx.z;

    const __bf16* W   = dir ? wb : wf;
    const float*  bih = dir ? bih_b : bih_f;
    float*        xg  = dir ? xg_b : xg_f;

    // A-matrix (16x32 bf16) lane mapping: row = m0+(lane&15);
    // elems 0..7 -> K = half*8 + {0..7}; elems 8..15 -> K = 16 + half*8 + {0..7}
    const int arow_fwd = m0 + (lane & 15);
    const int arow     = dir ? (TT - 1 - arow_fwd) : arow_fwd;  // reversed input for bwd dir
    const int ahalf    = (lane >> 4) * 8;

    // B-matrix (32x16 bf16) lane mapping: col = n0+(lane&15);
    // elems 0..7 -> K = (lane>>4)*8 + {0..7}; elems 8..15 -> K = 16 + (lane>>4)*8 + {0..7}; B[k][n] = W_ih[n][k]
    const int bcol  = n0 + (lane & 15);
    const int bhalf = (lane >> 4) * 16;

    const __bf16* arowp = xbf + (size_t)arow * DP;
    const __bf16* bcolp = W   + (size_t)bcol * DP;

    v8f c = {};
    for (int kk = 0; kk < DP; kk += 32) {
        if (kk + 32 < DP) {
            __builtin_prefetch(arowp + kk + 32, 0, 3);   // global_prefetch_b8
            __builtin_prefetch(bcolp + kk + 32, 0, 3);
        }
        v16bf a, b;
#pragma unroll
        for (int i = 0; i < 8; ++i) {
            a[i]     = arowp[kk + ahalf + i];
            a[8 + i] = arowp[kk + 16 + ahalf + i];
        }
#pragma unroll
        for (int i = 0; i < 16; ++i) {
            b[i] = bcolp[kk + ((i >> 3) << 4) + (bhalf >> 1) + (i & 7)];
        }
        c = __builtin_amdgcn_wmma_f32_16x16x32_bf16(
                /*neg_a=*/false, a, /*neg_b=*/false, b,
                /*c_mod=*/(short)0, c, /*reuse_a=*/false, /*reuse_b=*/false);
    }

    // C/D layout: elem e -> M = m0 + e + (lane>>4)*8, N = n0 + (lane&15)
    const int ncol  = n0 + (lane & 15);
    const int mbase = m0 + (lane >> 4) * 8;
    const float bias = bih[ncol];
#pragma unroll
    for (int e = 0; e < 8; ++e) {
        xg[(size_t)(mbase + e) * GG + ncol] = c[e] + bias;
    }
}

// ---------------- sequential LSTM recurrence, one block per direction ----------------
// W_hh pinned in VGPRs (128 packed bf16x2 per thread x 1024 threads = 512 KB),
// h broadcast through LDS; 2 barriers per step.
__global__ void __launch_bounds__(1024, 1)
lstm_kernel(const float* __restrict__ xg_f, const float* __restrict__ xg_b,
            const unsigned int* __restrict__ whh_pk,   // [2][GG][HH/2]
            const float* __restrict__ bhh_f, const float* __restrict__ bhh_b,
            float* __restrict__ hf, float* __restrict__ hb) {
    const int dir = blockIdx.x;
    const int gid = threadIdx.x;            // gate row 0..1023 (i,f,g,o blocks of 256)
    const float* xg  = dir ? xg_b : xg_f;
    const float* bhh = dir ? bhh_b : bhh_f;
    float*       hs  = dir ? hb : hf;

    const unsigned int* wrow = whh_pk + ((size_t)dir * GG + gid) * (HH / 2);
    unsigned int wreg[HH / 2];
#pragma unroll
    for (int j = 0; j < HH / 2; ++j) wreg[j] = wrow[j];

    __shared__ float hsh[HH];
    __shared__ float gsh[GG];
    float cst = 0.0f;
    if (gid < HH) hsh[gid] = 0.0f;
    const float bias = bhh[gid];
    __syncthreads();

    for (int t = 0; t < TT; ++t) {
        // register-resident rank-256 dot against broadcast h
        float acc = 0.0f;
        const float2* h2 = (const float2*)hsh;
#pragma unroll
        for (int j = 0; j < HH / 2; ++j) {
            unsigned int p = wreg[j];
            float w0 = __uint_as_float((p & 0xffffu) << 16);
            float w1 = __uint_as_float(p & 0xffff0000u);
            float2 hv = h2[j];
            acc = fmaf(w0, hv.x, acc);
            acc = fmaf(w1, hv.y, acc);
        }
        gsh[gid] = xg[(size_t)t * GG + gid] + bias + acc;
        __syncthreads();
        if (gid < HH) {
            float iv = sigmoidf_(gsh[gid]);
            float fv = sigmoidf_(gsh[HH + gid]);
            float gv = tanhf(gsh[2 * HH + gid]);
            float ov = sigmoidf_(gsh[3 * HH + gid]);
            cst = fv * cst + iv * gv;
            float h = ov * tanhf(cst);
            hsh[gid] = h;
            hs[(size_t)t * HH + gid] = h;
        }
        __syncthreads();
    }
}

// ---------------- output projection: feats = [hf, hb] @ W_out^T + b_out ----------------
__global__ void feats_kernel(const float* __restrict__ hf,
                             const float* __restrict__ hb,   // stored in reversed-time order
                             const float* __restrict__ Wout, // [KK][HIDN]
                             const float* __restrict__ bout,
                             float* __restrict__ feats) {    // [TT][KK]
    int t = blockIdx.x;
    int k = threadIdx.x;
    if (k >= KK) return;
    const float* hft = hf + (size_t)t * HH;
    const float* hbt = hb + (size_t)(TT - 1 - t) * HH;       // un-reverse backward dir
    const float* wr  = Wout + (size_t)k * HIDN;
    float acc = bout[k];
    for (int j = 0; j < HH; ++j) acc = fmaf(hft[j], wr[j], acc);
    for (int j = 0; j < HH; ++j) acc = fmaf(hbt[j], wr[HH + j], acc);
    feats[(size_t)t * KK + k] = acc;
}

// ---------------- Viterbi: sequential max-plus scan + backtrace ----------------
__global__ void viterbi_kernel(const float* __restrict__ feats,
                               const float* __restrict__ trans,  // [KK][KK], [to][from]
                               int* __restrict__ bptr,           // [TT][KK]
                               float* __restrict__ out) {        // [1 + TT]
    __shared__ float fv[KK];
    __shared__ float fvn[KK];
    int k = threadIdx.x;
    if (k < KK) fv[k] = (k == KSTART) ? 0.0f : -10000.0f;
    __syncthreads();
    for (int t = 0; t < TT; ++t) {
        if (k < KK) {
            float best = -3.4e38f; int bi = 0;
            for (int j = 0; j < KK; ++j) {
                float s = fv[j] + trans[k * KK + j];
                if (s > best) { best = s; bi = j; }   // strict > keeps first argmax
            }
            bptr[t * KK + k] = bi;
            fvn[k] = best + feats[(size_t)t * KK + k];
        }
        __syncthreads();
        if (k < KK) fv[k] = fvn[k];
        __syncthreads();
    }
    if (k == 0) {
        float best = -3.4e38f; int bi = 0;
        for (int j = 0; j < KK; ++j) {
            float s = fv[j] + trans[KSTOP * KK + j];
            if (s > best) { best = s; bi = j; }
        }
        out[0] = best;                       // path_score
        int tag = bi;
        for (int t = TT - 1; t >= 0; --t) {  // best_path emitted as floats
            out[1 + t] = (float)tag;
            tag = bptr[t * KK + tag];
        }
    }
}

// ---------------- host-side orchestration ----------------
extern "C" void kernel_launch(void* const* d_in, const int* in_sizes, int n_in,
                              void* d_out, int out_size, void* d_ws, size_t ws_size,
                              hipStream_t stream) {
    (void)in_sizes; (void)n_in; (void)out_size; (void)ws_size;

    const int*   sent  = (const int*)d_in[0];
    const float* E     = (const float*)d_in[1];
    const float* Wih_f = (const float*)d_in[2];
    const float* Whh_f = (const float*)d_in[3];
    const float* bih_f = (const float*)d_in[4];
    const float* bhh_f = (const float*)d_in[5];
    const float* Wih_b = (const float*)d_in[6];
    const float* Whh_b = (const float*)d_in[7];
    const float* bih_b = (const float*)d_in[8];
    const float* bhh_b = (const float*)d_in[9];
    const float* Wout  = (const float*)d_in[10];
    const float* bout  = (const float*)d_in[11];
    const float* trans = (const float*)d_in[12];
    float* out = (float*)d_out;

    char* ws = (char*)d_ws;
    size_t off = 0;
    auto alloc = [&](size_t bytes) -> void* {
        off = (off + 255) & ~(size_t)255;
        void* p = ws + off;
        off += bytes;
        return p;
    };
    unsigned short* xbf   = (unsigned short*)alloc((size_t)TT * DP * 2);
    unsigned short* wihbf = (unsigned short*)alloc((size_t)2 * GG * DP * 2);
    unsigned int*   whhpk = (unsigned int*)alloc((size_t)2 * GG * (HH / 2) * 4);
    float* xgf   = (float*)alloc((size_t)TT * GG * 4);
    float* xgb   = (float*)alloc((size_t)TT * GG * 4);
    float* hf    = (float*)alloc((size_t)TT * HH * 4);
    float* hb    = (float*)alloc((size_t)TT * HH * 4);
    float* feats = (float*)alloc((size_t)TT * KK * 4);
    int*   bptr  = (int*)alloc((size_t)TT * KK * 4);

    {
        int n = TT * DP;
        prep_x_kernel<<<(n + 255) / 256, 256, 0, stream>>>(sent, E, xbf);
    }
    {
        int n = 2 * GG * DP;
        prep_wih_kernel<<<(n + 255) / 256, 256, 0, stream>>>(Wih_f, Wih_b, wihbf);
    }
    {
        int n = 2 * GG * (HH / 2);
        prep_whh_kernel<<<(n + 255) / 256, 256, 0, stream>>>(Whh_f, Whh_b, whhpk);
    }
    {
        dim3 grid(TT / 16, GG / 16 / 4, 2);  // (128, 16, 2), 4 waves/block
        xg_gemm_kernel<<<grid, 128, 0, stream>>>(
            (const __bf16*)xbf,
            (const __bf16*)wihbf,
            (const __bf16*)(wihbf + (size_t)GG * DP),
            bih_f, bih_b, xgf, xgb);
    }
    lstm_kernel<<<2, 1024, 0, stream>>>(xgf, xgb, whhpk, bhh_f, bhh_b, hf, hb);
    feats_kernel<<<TT, 64, 0, stream>>>(hf, hb, Wout, bout, feats);
    viterbi_kernel<<<1, 64, 0, stream>>>(feats, trans, bptr, out);
}
